// FluxIPAttnProcessor_24378234372282
// MI455X (gfx1250) — matrix operand order repaired and checked by reading, not compiled, not run on hardware
//
#include <hip/hip_runtime.h>
#include <hip/hip_bf16.h>
#include <math.h>

// Problem constants (from reference)
#define S_LEN 2048
#define NH    24
#define HD    128
#define HIDN  3072
#define RDIM  64
#define IPSL  512
#define IPDN  1152
#define EPSV  1e-6f
#define INV_SQRT_D 0.08838834764831845f  // 128^-0.5

typedef __attribute__((ext_vector_type(16))) __bf16         v16bf;
typedef __attribute__((ext_vector_type(8)))  float          v8f;
typedef __attribute__((ext_vector_type(8)))  unsigned short v8us;
typedef __attribute__((ext_vector_type(16))) unsigned short v16us;

__device__ __forceinline__ unsigned short f2bf_bits(float f) {
    union { float f; unsigned u; } c; c.f = f;
    unsigned u = c.u;
    u += 0x7fffu + ((u >> 16) & 1u);   // round-to-nearest-even
    return (unsigned short)(u >> 16);
}

// 16 contiguous bf16 (B-fragment: K = half*16 + 0..15, caller offsets ptr)
__device__ __forceinline__ v16bf load_bf16x16(const unsigned short* p) {
    v8us a = *(const v8us*)p;
    v8us b = *(const v8us*)(p + 8);
    v16us u;
#pragma unroll
    for (int i = 0; i < 8; ++i) { u[i] = a[i]; u[8 + i] = b[i]; }
    return __builtin_bit_cast(v16bf, u);
}

// A-fragment: 8 halfs at p, 8 halfs at p+16 (K = base..+7, base+16..+23)
__device__ __forceinline__ v16bf load_bf16_agap(const unsigned short* p) {
    v8us a = *(const v8us*)p;
    v8us b = *(const v8us*)(p + 16);
    v16us u;
#pragma unroll
    for (int i = 0; i < 8; ++i) { u[i] = a[i]; u[8 + i] = b[i]; }
    return __builtin_bit_cast(v16bf, u);
}

// ---- CDNA5 async mem->LDS (ASYNCcnt path); offset applies to mem AND lds addr
__device__ __forceinline__ void gl_async_b128(unsigned lds, const unsigned short* g) {
    asm volatile("global_load_async_to_lds_b128 %0, %1, off"
                 :: "v"(lds), "v"((unsigned long long)(size_t)g) : "memory");
}
__device__ __forceinline__ void gl_async_b128_o16(unsigned lds, const unsigned short* g) {
    asm volatile("global_load_async_to_lds_b128 %0, %1, off offset:16"
                 :: "v"(lds), "v"((unsigned long long)(size_t)g) : "memory");
}
__device__ __forceinline__ void gl_async_b128_o32(unsigned lds, const unsigned short* g) {
    asm volatile("global_load_async_to_lds_b128 %0, %1, off offset:32"
                 :: "v"(lds), "v"((unsigned long long)(size_t)g) : "memory");
}
__device__ __forceinline__ void gl_async_b128_o48(unsigned lds, const unsigned short* g) {
    asm volatile("global_load_async_to_lds_b128 %0, %1, off offset:48"
                 :: "v"(lds), "v"((unsigned long long)(size_t)g) : "memory");
}

// ---------------------------------------------------------------------------
// fp32 -> bf16 bulk convert (n must be a multiple of 256)
// ---------------------------------------------------------------------------
__launch_bounds__(256)
__global__ void cvt_bf16_kernel(const float* __restrict__ in, unsigned short* __restrict__ out) {
    const size_t i = (size_t)blockIdx.x * 256 + threadIdx.x;
    out[i] = f2bf_bits(in[i]);
}

// ---------------------------------------------------------------------------
// GEMM:  C[M,N] = A[M,K] @ W[N,K]^T + bias[N]   (bf16 in, f32 WMMA acc, f32 out)
// Block = 256 threads = 8 waves; WG tile 64(M) x 256(N); wave tile 16 x 128.
// A/B tiles double-buffered in LDS via global_load_async_to_lds_b128.
// Steady-state: issue s+1, s_wait_asynccnt 5, barrier, 18 ds_b128 loads into
// distinct regs, ONE s_wait_dscnt, 8 back-to-back WMMAs, barrier, flip.
// ---------------------------------------------------------------------------
__device__ __forceinline__ void gemm_issue(unsigned la, unsigned lb,
                                           const unsigned short* ag, const unsigned short* bg) {
    gl_async_b128    (la, ag);
    gl_async_b128    (lb, bg);
    gl_async_b128_o16(lb, bg);
    gl_async_b128_o32(lb, bg);
    gl_async_b128_o48(lb, bg);
}

__device__ __forceinline__ void gemm_step(const unsigned short* Al, const unsigned short* Bl,
                                          int wm, int wn, int l, int half, v8f acc[8]) {
    v16bf af = load_bf16x16(&Al[(wm * 16 + l) * 32 + half * 16]);
    v16bf bfr[8];
#pragma unroll
    for (int j = 0; j < 8; ++j)
        bfr[j] = load_bf16x16(&Bl[(wn * 128 + j * 16 + l) * 32 + half * 16]);
#pragma unroll
    for (int j = 0; j < 8; ++j)
        acc[j] = __builtin_amdgcn_wmma_f32_16x16x32_bf16(
            false, af, false, bfr[j], (short)0, acc[j], false, false);
}

__launch_bounds__(256)
__global__ void gemm_bb(const unsigned short* __restrict__ A, const unsigned short* __restrict__ W,
                        const float* __restrict__ bias, float* __restrict__ C,
                        int M, int N, int K) {
    __shared__ __align__(16) unsigned short Abuf[2][64 * 32];    //  4 KB x2
    __shared__ __align__(16) unsigned short Bbuf[2][256 * 32];   // 16 KB x2

    const int tid  = threadIdx.x;
    const int lane = tid & 31;
    const int wave = tid >> 5;
    const int half = lane >> 4;
    const int l    = lane & 15;
    const int wm   = wave >> 1;          // 0..3  -> M sub-tile
    const int wn   = wave & 1;           // 0..1  -> N sub-tile
    const int mblk = blockIdx.y * 64;
    const int nblk = blockIdx.x * 256;

    // per-thread staging coordinates
    const int arow  = tid >> 2;          // 0..63
    const int acol8 = (tid & 3) * 8;     // 0,8,16,24
    const unsigned ldsA0 = (unsigned)(size_t)(void*)&Abuf[0][arow * 32 + acol8];
    const unsigned ldsA1 = (unsigned)(size_t)(void*)&Abuf[1][arow * 32 + acol8];
    const unsigned ldsB0 = (unsigned)(size_t)(void*)&Bbuf[0][tid * 32];
    const unsigned ldsB1 = (unsigned)(size_t)(void*)&Bbuf[1][tid * 32];
    const unsigned short* Ag = A + (size_t)(mblk + arow) * K + acol8;
    const unsigned short* Bg = W + (size_t)(nblk + tid) * K;

    // prologue: stage 0 into buffer 0
    gemm_issue(ldsA0, ldsB0, Ag, Bg);

    v8f acc[8] = {};
    int buf = 0;
    // steady state (all but last K-step): unconditional issue + wait 5
    for (int kk = 0; kk < K - 32; kk += 32) {
        const int nxt = kk + 32;
        gemm_issue(buf ? ldsA0 : ldsA1, buf ? ldsB0 : ldsB1, Ag + nxt, Bg + nxt);
        asm volatile("s_wait_asynccnt 0x5" ::: "memory");
        __syncthreads();
        gemm_step(buf ? Abuf[1] : Abuf[0], buf ? Bbuf[1] : Bbuf[0], wm, wn, l, half, acc);
        __syncthreads();
        buf ^= 1;
    }
    // final K-step
    asm volatile("s_wait_asynccnt 0x0" ::: "memory");
    __syncthreads();
    gemm_step(buf ? Abuf[1] : Abuf[0], buf ? Bbuf[1] : Bbuf[0], wm, wn, l, half, acc);

#pragma unroll
    for (int j = 0; j < 8; ++j) {
#pragma unroll
        for (int r = 0; r < 8; ++r) {
            const int row = mblk + wm * 16 + r + half * 8;
            const int col = nblk + wn * 128 + j * 16 + l;
            C[(size_t)row * N + col] = acc[j][r] + bias[col];
        }
    }
}

// ---------------------------------------------------------------------------
// Prep: per (s, head) row — RMSNorm(q)*scale*D^-0.5 -> qn (bf16),
//        RoPE(q)->qr (bf16), RoPE(k)->kr (bf16).  128 threads = one row.
// ---------------------------------------------------------------------------
__launch_bounds__(128)
__global__ void prep_qk(const float* __restrict__ Q, const float* __restrict__ Kin,
                        const float* __restrict__ sinp, const float* __restrict__ cosp,
                        const float* __restrict__ q_scale,
                        unsigned short* __restrict__ qr, unsigned short* __restrict__ kr,
                        unsigned short* __restrict__ qn) {
    __shared__ float qs[HD], ks[HD], red[4];
    const int s = blockIdx.x, h = blockIdx.y, d = threadIdx.x;
    const size_t idx = (size_t)s * HIDN + (size_t)h * HD + d;
    const float qv = Q[idx], kv = Kin[idx];
    qs[d] = qv; ks[d] = kv;
    float ss = qv * qv;
#pragma unroll
    for (int m = 16; m >= 1; m >>= 1) ss += __shfl_xor(ss, m, 32);
    if ((d & 31) == 0) red[d >> 5] = ss;
    __syncthreads();
    const float tot = red[0] + red[1] + red[2] + red[3];
    const float rms = sqrtf(tot / (float)HD);
    qn[idx] = f2bf_bits(q_scale[d] * (qv / (rms + EPSV)) * INV_SQRT_D);
    float qo, ko;
    if (d < RDIM) {
        const float sn = sinp[s * RDIM + d], cs = cosp[s * RDIM + d];
        const float qrot = (d < 32) ? -qs[d + 32] : qs[d - 32];
        const float krot = (d < 32) ? -ks[d + 32] : ks[d - 32];
        qo = qv * cs + qrot * sn;
        ko = kv * cs + krot * sn;
    } else { qo = qv; ko = kv; }
    qr[idx] = f2bf_bits(qo);
    kr[idx] = f2bf_bits(ko);
}

__launch_bounds__(128)
__global__ void prep_kip(const float* __restrict__ Kip, const float* __restrict__ k_scale,
                         unsigned short* __restrict__ kipn) {
    __shared__ float red[4];
    const int s = blockIdx.x, h = blockIdx.y, d = threadIdx.x;
    const size_t idx = (size_t)s * HIDN + (size_t)h * HD + d;
    const float kv = Kip[idx];
    float ss = kv * kv;
#pragma unroll
    for (int m = 16; m >= 1; m >>= 1) ss += __shfl_xor(ss, m, 32);
    if ((d & 31) == 0) red[d >> 5] = ss;
    __syncthreads();
    const float tot = red[0] + red[1] + red[2] + red[3];
    const float rms = sqrtf(tot / (float)HD);
    kipn[idx] = f2bf_bits(k_scale[d] * (kv / (rms + EPSV)) * INV_SQRT_D);
}

__launch_bounds__(256)
__global__ void add_bf16_kernel(const float* __restrict__ a, const float* __restrict__ b,
                                unsigned short* __restrict__ o) {
    const size_t i = (size_t)blockIdx.x * 256 + threadIdx.x;
    o[i] = f2bf_bits(a[i] + b[i]);
}

// ---------------------------------------------------------------------------
// Flash attention (one softmax branch): Out[s, h*D+d] = softmax(Q K^T) V.
// Block = 256 thr = 8 waves; each wave owns 16 query rows; keys in tiles of 32.
// K-tile via global_load_async_to_lds_b128; V transposed through VGPRs;
// P restriped D->A layout via wave-private LDS; B-frags preloaded in groups.
// ---------------------------------------------------------------------------
__launch_bounds__(256)
__global__ void attn_wmma(const unsigned short* __restrict__ Qb,
                          const unsigned short* __restrict__ Kb,
                          const unsigned short* __restrict__ Vb,
                          float* __restrict__ Out, int T) {
    __shared__ __align__(16) unsigned short Kt[32 * 128];   // [key][d]
    __shared__ __align__(16) unsigned short Vt[128 * 32];   // [d][key] (transposed)
    __shared__ __align__(16) unsigned short Pst[8 * 16 * 32];

    const int tid  = threadIdx.x;
    const int lane = tid & 31, wave = tid >> 5;
    const int half = lane >> 4, l = lane & 15;
    const int h  = blockIdx.y;
    const int q0 = blockIdx.x * 128 + wave * 16;
    const size_t hoff = (size_t)h * HD;

    // Q fragments for the 4 K-dim steps of 32 (A layout)
    v16bf qf[4];
    {
        const unsigned short* qrow = Qb + (size_t)(q0 + l) * HIDN + hoff;
#pragma unroll
        for (int kd = 0; kd < 4; ++kd)
            qf[kd] = load_bf16x16(qrow + kd * 32 + half * 16);
    }

    v8f acc[8] = {};
    float mrow[8], lrow[8];
#pragma unroll
    for (int r = 0; r < 8; ++r) { mrow[r] = -1e30f; lrow[r] = 0.f; }

    const int row  = tid >> 3;         // 0..31 (key within tile)
    const int coff = (tid & 7) * 16;   // 0..112 (d offset)
    const unsigned ldsK = (unsigned)(size_t)(void*)&Kt[row * 128 + coff];

    for (int t0 = 0; t0 < T; t0 += 32) {
        // ---- K tile: async DMA to LDS (32 bytes/thread)
        {
            const unsigned short* kp = Kb + (size_t)(t0 + row) * HIDN + hoff + coff;
            gl_async_b128    (ldsK, kp);
            gl_async_b128_o16(ldsK, kp);
        }
        // ---- V tile: load + transpose through VGPRs
        {
            const unsigned short* vp = Vb + (size_t)(t0 + row) * HIDN + hoff + coff;
            v8us c0 = *(const v8us*)vp;
            v8us c1 = *(const v8us*)(vp + 8);
#pragma unroll
            for (int i = 0; i < 8; ++i) Vt[(coff + i) * 32 + row]     = c0[i];
#pragma unroll
            for (int i = 0; i < 8; ++i) Vt[(coff + 8 + i) * 32 + row] = c1[i];
        }
        asm volatile("s_wait_asynccnt 0x0" ::: "memory");
        __syncthreads();

        // ---- scores: S0 = keys t0..t0+15, S1 = keys t0+16..t0+31
        //      process 2 K-steps per group with 4 preloaded B-fragments
        v8f s0 = {}, s1 = {};
#pragma unroll
        for (int kd = 0; kd < 4; kd += 2) {
            v16bf b[4];
            b[0] = load_bf16x16(&Kt[l * 128        + kd * 32       + half * 16]);
            b[1] = load_bf16x16(&Kt[(l + 16) * 128 + kd * 32       + half * 16]);
            b[2] = load_bf16x16(&Kt[l * 128        + (kd + 1) * 32 + half * 16]);
            b[3] = load_bf16x16(&Kt[(l + 16) * 128 + (kd + 1) * 32 + half * 16]);
            s0 = __builtin_amdgcn_wmma_f32_16x16x32_bf16(false, qf[kd],     false, b[0], (short)0, s0, false, false);
            s1 = __builtin_amdgcn_wmma_f32_16x16x32_bf16(false, qf[kd],     false, b[1], (short)0, s1, false, false);
            s0 = __builtin_amdgcn_wmma_f32_16x16x32_bf16(false, qf[kd + 1], false, b[2], (short)0, s0, false, false);
            s1 = __builtin_amdgcn_wmma_f32_16x16x32_bf16(false, qf[kd + 1], false, b[3], (short)0, s1, false, false);
        }

        // ---- online softmax (row = q0 + r + 8*half; columns across 16 lanes)
#pragma unroll
        for (int r = 0; r < 8; ++r) {
            float mx = fmaxf(s0[r], s1[r]);
#pragma unroll
            for (int msk = 8; msk >= 1; msk >>= 1) mx = fmaxf(mx, __shfl_xor(mx, msk, 32));
            const float nm   = fmaxf(mrow[r], mx);
            const float corr = __expf(mrow[r] - nm);
            const float p0 = __expf(s0[r] - nm);
            const float p1 = __expf(s1[r] - nm);
            s0[r] = p0; s1[r] = p1;
            float rs = p0 + p1;
#pragma unroll
            for (int msk = 8; msk >= 1; msk >>= 1) rs += __shfl_xor(rs, msk, 32);
            lrow[r] = lrow[r] * corr + rs;
            mrow[r] = nm;
#pragma unroll
            for (int j = 0; j < 8; ++j) acc[j][r] *= corr;
        }

        // ---- restripe P (D-layout -> A-layout) through wave-private LDS
        unsigned short* pw = &Pst[wave * 512];
#pragma unroll
        for (int r = 0; r < 8; ++r) {
            const int rr = r + half * 8;
            pw[rr * 32 + l]      = f2bf_bits(s0[r]);
            pw[rr * 32 + 16 + l] = f2bf_bits(s1[r]);
        }
        v16bf pf = load_bf16x16(&pw[l * 32 + half * 16]);

        // ---- PV: 8 output d-tiles, V-fragments preloaded in groups of 4
#pragma unroll
        for (int jg = 0; jg < 2; ++jg) {
            v16bf vf[4];
#pragma unroll
            for (int j = 0; j < 4; ++j)
                vf[j] = load_bf16x16(&Vt[((jg * 4 + j) * 16 + l) * 32 + half * 16]);
#pragma unroll
            for (int j = 0; j < 4; ++j)
                acc[jg * 4 + j] = __builtin_amdgcn_wmma_f32_16x16x32_bf16(
                    false, pf, false, vf[j], (short)0, acc[jg * 4 + j], false, false);
        }
        __syncthreads();
    }

    // ---- epilogue: divide by softmax denominator, write fp32
#pragma unroll
    for (int j = 0; j < 8; ++j)
#pragma unroll
        for (int r = 0; r < 8; ++r) {
            const int srow = q0 + r + half * 8;
            Out[(size_t)srow * HIDN + hoff + j * 16 + l] = acc[j][r] / lrow[r];
        }
}

// ---------------------------------------------------------------------------
extern "C" void kernel_launch(void* const* d_in, const int* in_sizes, int n_in,
                              void* d_out, int out_size, void* d_ws, size_t ws_size,
                              hipStream_t stream) {
    (void)in_sizes; (void)n_in; (void)out_size; (void)ws_size;
    const float* hidden = (const float*)d_in[0];
    const float* iph    = (const float*)d_in[1];
    const float* sinp   = (const float*)d_in[2];
    const float* cosp   = (const float*)d_in[3];
    const float* Wq = (const float*)d_in[4];  const float* bq = (const float*)d_in[5];
    const float* Wk = (const float*)d_in[6];  const float* bk = (const float*)d_in[7];
    const float* Wv = (const float*)d_in[8];  const float* bv = (const float*)d_in[9];
    const float* Wo = (const float*)d_in[10]; const float* bo = (const float*)d_in[11];
    const float* Wkip = (const float*)d_in[12]; const float* bkip = (const float*)d_in[13];
    const float* Wvip = (const float*)d_in[14]; const float* bvip = (const float*)d_in[15];
    const float* qscale = (const float*)d_in[16];
    const float* kscale = (const float*)d_in[17];

    const size_t SH  = (size_t)S_LEN * HIDN;   // 6,291,456
    const size_t PH  = (size_t)IPSL  * HIDN;   // 1,572,864
    const size_t IH  = (size_t)IPSL  * IPDN;   //   589,824
    const size_t WW  = (size_t)HIDN * HIDN;    // 9,437,184
    const size_t WIP = (size_t)HIDN * IPDN;    // 3,538,944

    // ---- workspace layout
    float* Qf   = (float*)d_ws;
    float* Kf   = Qf + SH;
    float* Vf   = Kf + SH;
    float* Kipf = Vf + SH;
    float* Vipf = Kipf + PH;
    unsigned short* qr    = (unsigned short*)(Vipf + PH);
    unsigned short* kr    = qr + SH;
    unsigned short* qn    = kr + SH;
    unsigned short* vb    = qn + SH;
    unsigned short* kipn  = vb + SH;
    unsigned short* vipb  = kipn + PH;
    unsigned short* hb    = vipb + PH;   // hidden bf16
    unsigned short* ipb   = hb + SH;     // ip_hidden bf16
    unsigned short* attnb = ipb + IH;    // (attn1+attn2) bf16
    unsigned short* Wqb   = attnb + SH;
    unsigned short* Wkb   = Wqb + WW;
    unsigned short* Wvb   = Wkb + WW;
    unsigned short* Wipkb = Wvb + WW;
    unsigned short* Wipvb = Wipkb + WIP;
    unsigned short* Wob   = Wqb;         // alias: converted after QKV GEMMs (stream order)
    float* attn1 = Qf;                   // alias: Q fp32 dead after prep_qk
    float* attn2 = Kf;                   // alias: K fp32 dead after prep_qk

    const dim3 blk256(256), blk128(128);
    const dim3 gQKV(HIDN / 256, S_LEN / 64);   // 12 x 32
    const dim3 gIP (HIDN / 256, IPSL  / 64);   // 12 x 8
    const dim3 gAtt(S_LEN / 128, NH);          // 16 x 24

    // ---- one-time bf16 conversion of GEMM operands
    cvt_bf16_kernel<<<SH  / 256, blk256, 0, stream>>>(hidden, hb);
    cvt_bf16_kernel<<<IH  / 256, blk256, 0, stream>>>(iph,   ipb);
    cvt_bf16_kernel<<<WW  / 256, blk256, 0, stream>>>(Wq,    Wqb);
    cvt_bf16_kernel<<<WW  / 256, blk256, 0, stream>>>(Wk,    Wkb);
    cvt_bf16_kernel<<<WW  / 256, blk256, 0, stream>>>(Wv,    Wvb);
    cvt_bf16_kernel<<<WIP / 256, blk256, 0, stream>>>(Wkip,  Wipkb);
    cvt_bf16_kernel<<<WIP / 256, blk256, 0, stream>>>(Wvip,  Wipvb);

    // ---- QKV + IP projections (LDS-staged async WMMA GEMMs)
    gemm_bb<<<gQKV, blk256, 0, stream>>>(hb, Wqb, bq, Qf, S_LEN, HIDN, HIDN);
    gemm_bb<<<gQKV, blk256, 0, stream>>>(hb, Wkb, bk, Kf, S_LEN, HIDN, HIDN);
    gemm_bb<<<gQKV, blk256, 0, stream>>>(hb, Wvb, bv, Vf, S_LEN, HIDN, HIDN);
    gemm_bb<<<gIP,  blk256, 0, stream>>>(ipb, Wipkb, bkip, Kipf, IPSL, HIDN, IPDN);
    gemm_bb<<<gIP,  blk256, 0, stream>>>(ipb, Wipvb, bvip, Vipf, IPSL, HIDN, IPDN);

    // Wo bf16 into Wq's slot (safe: stream-ordered after QKV GEMMs)
    cvt_bf16_kernel<<<WW / 256, blk256, 0, stream>>>(Wo, Wob);

    // ---- elementwise prep -> bf16 attention operands
    prep_qk <<<dim3(S_LEN, NH), blk128, 0, stream>>>(Qf, Kf, sinp, cosp, qscale, qr, kr, qn);
    prep_kip<<<dim3(IPSL,  NH), blk128, 0, stream>>>(Kipf, kscale, kipn);
    cvt_bf16_kernel<<<SH / 256, blk256, 0, stream>>>(Vf,   vb);
    cvt_bf16_kernel<<<PH / 256, blk256, 0, stream>>>(Vipf, vipb);

    // ---- dual attention branches (flash, WMMA)
    attn_wmma<<<gAtt, blk256, 0, stream>>>(qr, kr,   vb,   attn1, S_LEN);
    attn_wmma<<<gAtt, blk256, 0, stream>>>(qn, kipn, vipb, attn2, IPSL);
    add_bf16_kernel<<<SH / 256, blk256, 0, stream>>>(attn1, attn2, attnb);

    // ---- output projection -> d_out (fp32)
    gemm_bb<<<gQKV, blk256, 0, stream>>>(attnb, Wob, bo, (float*)d_out, S_LEN, HIDN, HIDN);
}
